// Qwen3Attention_13383118095061
// MI455X (gfx1250) — matrix operand order repaired and checked by reading, not compiled, not run on hardware
//
#include <hip/hip_runtime.h>
#include <hip/hip_bf16.h>

// ---------------------------------------------------------------------------
// Qwen3-style attention layer for MI455X (gfx1250), wave32 + WMMA bf16.
//   B=2, S=2048, H=4096, NUM_HEADS=32, NUM_KV_HEADS=8, HEAD_DIM=128
// Pipeline:
//   1) f32 -> bf16 converts (hidden, wqkv, wo)
//   2) GEMM1: qkv[4096,6144] = hidden[4096,4096] x wqkv^T   (WMMA bf16)
//   3) per-head RMSNorm + RoPE -> Qb, Kb (row-major d), VT (transposed [d][s])
//   4) flash attention (WMMA bf16, online softmax)          -> attnb bf16
//   5) GEMM2: out[4096,4096] = attnb x wo^T                 (WMMA bf16, f32 out)
//
// GEMM uses 64x64 wave tiles (16 WMMA per 16 b128 loads per K-step) in a
// 128x256 block tile; all dims divide evenly so there are no bounds checks.
// ---------------------------------------------------------------------------

typedef __bf16 bf16;
typedef __attribute__((ext_vector_type(16))) __bf16 v16bf;
typedef __attribute__((ext_vector_type(8)))  __bf16 v8bf;
typedef __attribute__((ext_vector_type(8)))  float  v8f;

#define BB   2
#define SS   2048
#define HH   4096
#define NH   32
#define NKV  8
#define HD   128
#define QKVO 6144
#define MTOT (BB * SS)           // 4096
#define GRP  (NH / NKV)          // 4

static __device__ __forceinline__ v8f splat8(float x) {
  v8f r;
#pragma unroll
  for (int i = 0; i < 8; ++i) r[i] = x;
  return r;
}

static __device__ __forceinline__ v8f wmma_bf16(v16bf a, v16bf b, v8f c) {
  // D = A(16x32) x B(32x16) + C(16x16), fp32 accumulate
  return __builtin_amdgcn_wmma_f32_16x16x32_bf16(
      false, a, false, b, (short)0, c, false, false);
}

// A fragment (16x32 bf16): lane<16 -> row=lane, K = {0..15};
// lane>=16 -> row=lane-16, K = {16..31}.  base = &A[row0][k0], ldk in elems.
static __device__ __forceinline__ v16bf load_a_frag(const bf16* base, int ldk, int lane) {
  const bf16* p = base + (size_t)(lane & 15) * ldk + ((lane >> 4) << 4);
  v8bf lo = *(const v8bf*)(p);
  v8bf hi = *(const v8bf*)(p + 8);
  v16bf r;
#pragma unroll
  for (int i = 0; i < 8; ++i) { r[i] = lo[i]; r[i + 8] = hi[i]; }
  return r;
}

// B fragment (32x16 bf16), source stored row-major [n][k] (k contiguous):
// col n = lane&15, elems i -> k = (lane>=16 ? 16 : 0) + i.  base = &Bsrc[n0][k0].
static __device__ __forceinline__ v16bf load_b_frag(const bf16* base, int ldk, int lane) {
  const bf16* p = base + (size_t)(lane & 15) * ldk + ((lane >> 4) << 4);
  return *(const v16bf*)(p);
}

// ---------------------------------------------------------------------------
// f32 -> bf16 convert
// ---------------------------------------------------------------------------
__global__ __launch_bounds__(256) void f32_to_bf16_k(const float* __restrict__ x,
                                                     bf16* __restrict__ y, size_t n) {
  size_t i = (size_t)blockIdx.x * blockDim.x + threadIdx.x;
  if (i < n) y[i] = (bf16)x[i];
}

// ---------------------------------------------------------------------------
// GEMM: C[M,N](f32) = A[M,K](bf16) x Bw[N,K](bf16)^T
// block = 256 threads (8 waves); block tile 128x256; wave tile 64x64
// (4x4 WMMA tiles -> 16 v_wmma per K-step of 32).
// ---------------------------------------------------------------------------
__global__ __launch_bounds__(256) void gemm_bf16_nt(const bf16* __restrict__ A,
                                                    const bf16* __restrict__ Bw,
                                                    float* __restrict__ C,
                                                    int M, int N, int K) {
  const int lane = threadIdx.x & 31;
  const int wid  = threadIdx.x >> 5;
  const int wm   = wid >> 2;                  // 0..1
  const int wn   = wid & 3;                   // 0..3
  const int m0   = blockIdx.y * 128 + wm * 64;
  const int n0   = blockIdx.x * 256 + wn * 64;
  const int rowA = lane & 15;

  v8f acc[4][4];
#pragma unroll
  for (int i = 0; i < 4; ++i)
#pragma unroll
    for (int j = 0; j < 4; ++j) acc[i][j] = splat8(0.0f);

  for (int k = 0; k < K; k += 32) {
    v16bf a[4], b[4];
#pragma unroll
    for (int i = 0; i < 4; ++i)
      a[i] = load_a_frag(A + (size_t)(m0 + i * 16) * K + k, K, lane);
#pragma unroll
    for (int j = 0; j < 4; ++j)
      b[j] = load_b_frag(Bw + (size_t)(n0 + j * 16) * K + k, K, lane);

    if (k + 32 < K) {  // prefetch next K tile into WGP-level caches
      __builtin_prefetch((const void*)(A  + (size_t)(m0 + rowA) * K + k + 32), 0, 3);
      __builtin_prefetch((const void*)(A  + (size_t)(m0 + 32 + rowA) * K + k + 32), 0, 3);
      __builtin_prefetch((const void*)(Bw + (size_t)(n0 + rowA) * K + k + 32), 0, 3);
      __builtin_prefetch((const void*)(Bw + (size_t)(n0 + 32 + rowA) * K + k + 32), 0, 3);
    }

#pragma unroll
    for (int i = 0; i < 4; ++i)
#pragma unroll
      for (int j = 0; j < 4; ++j)
        acc[i][j] = wmma_bf16(a[i], b[j], acc[i][j]);
  }

  const int rbase = (lane >> 4) << 3;  // C row = r + 8*(lane>=16)
  const int cn    = lane & 15;
#pragma unroll
  for (int i = 0; i < 4; ++i)
#pragma unroll
    for (int j = 0; j < 4; ++j)
#pragma unroll
      for (int r = 0; r < 8; ++r)
        C[(size_t)(m0 + i * 16 + rbase + r) * N + (n0 + j * 16 + cn)] = acc[i][j][r];
}

// ---------------------------------------------------------------------------
// Per-(token, head) RMSNorm + RoPE; splits qkv into Qb, Kb, VT(bf16).
// grid = (B*S, 48), block = 128 (one head vector per block).
//   head 0..31  -> Q: rmsnorm(q_norm_w) + rope -> Qb[b][h][s][d]
//   head 32..39 -> K: rmsnorm(k_norm_w) + rope -> Kb[b][kh][s][d]
//   head 40..47 -> V: convert, transposed       -> VT[b][vh][d][s]
// ---------------------------------------------------------------------------
__global__ __launch_bounds__(128) void qkv_post_k(const float* __restrict__ qkv,
                                                  const int*   __restrict__ pos,
                                                  const float* __restrict__ qnw,
                                                  const float* __restrict__ knw,
                                                  bf16* __restrict__ Qb,
                                                  bf16* __restrict__ Kb,
                                                  bf16* __restrict__ VT) {
  const int token = blockIdx.x;
  const int head  = blockIdx.y;
  const int d     = threadIdx.x;
  const int b     = token >> 11;       // /S
  const int s     = token & (SS - 1);

  const float v = qkv[(size_t)token * QKVO + head * HD + d];

  if (head >= NH + NKV) {  // V head: straight transpose-store
    const int vh = head - (NH + NKV);
    VT[((size_t)(b * NKV + vh) * HD + d) * SS + s] = (bf16)v;
    return;
  }

  __shared__ float red[4];
  __shared__ float xbuf[HD];

  // mean of squares over 128
  float sq = v * v;
#pragma unroll
  for (int m = 1; m < 32; m <<= 1) sq += __shfl_xor(sq, m, 32);
  const int lane = threadIdx.x & 31, wv = threadIdx.x >> 5;
  if (lane == 0) red[wv] = sq;
  __syncthreads();
  const float var = (red[0] + red[1] + red[2] + red[3]) * (1.0f / 128.0f);
  const float rs  = rsqrtf(var + 1e-6f);
  const float* w  = (head < NH) ? qnw : knw;
  const float xn  = v * rs * w[d];
  xbuf[d] = xn;
  __syncthreads();

  const float p = (float)pos[s];
  float out;
  if (d < 64) {
    const float inv = __powf(10000.0f, -(float)d / 64.0f);
    const float ang = p * inv;
    out = xn * __cosf(ang) - xbuf[d + 64] * __sinf(ang);
  } else {
    const int   j   = d - 64;
    const float inv = __powf(10000.0f, -(float)j / 64.0f);
    const float ang = p * inv;
    out = xn * __cosf(ang) + xbuf[j] * __sinf(ang);
  }

  if (head < NH)
    Qb[((size_t)(b * NH + head) * SS + s) * HD + d] = (bf16)out;
  else {
    const int kh = head - NH;
    Kb[((size_t)(b * NKV + kh) * SS + s) * HD + d] = (bf16)out;
  }
}

// ---------------------------------------------------------------------------
// Flash attention (causal, GQA group=4). grid = (S/64, B*NH), block = 128.
// Each wave owns 16 query rows; Q fragments in registers; K-loop in 32-chunks:
//   8 WMMAs for S = Q K^T, online softmax, 8 WMMAs for O += P V.
// P converted C-layout -> A-layout via per-wave LDS scratch (in-order LDS).
// ---------------------------------------------------------------------------
__global__ __launch_bounds__(128) void flash_attn_k(const bf16* __restrict__ Qb,
                                                    const bf16* __restrict__ Kb,
                                                    const bf16* __restrict__ VT,
                                                    bf16* __restrict__ attnb) {
  const int lane = threadIdx.x & 31;
  const int wid  = threadIdx.x >> 5;
  const int qt   = blockIdx.x;          // 0..S/64-1
  const int bh   = blockIdx.y;          // 0..B*NH-1
  const int b    = bh >> 5;
  const int h    = bh & 31;
  const int kvh  = h >> 2;              // GRP = 4
  const int q0   = qt * 64 + wid * 16;
  const float scale = 0.08838834764831845f;  // 1/sqrt(128)

  __shared__ bf16 pscr[4][16 * 32];
  bf16* scr = pscr[wid];

  const bf16* Qbase = Qb + ((size_t)(b * NH  + h)   * SS + q0) * HD;
  const bf16* Kbase = Kb + ((size_t)(b * NKV + kvh) * SS) * HD;
  const bf16* Vbase = VT + ((size_t)(b * NKV + kvh) * HD) * SS;

  v16bf qf[4];
#pragma unroll
  for (int c = 0; c < 4; ++c) qf[c] = load_a_frag(Qbase + c * 32, HD, lane);

  v8f mi = splat8(-3.0e38f);
  v8f li = splat8(0.0f);
  v8f oacc[8];
#pragma unroll
  for (int t = 0; t < 8; ++t) oacc[t] = splat8(0.0f);

  const int rbase = (lane >> 4) << 3;   // C row = r + rbase
  const int cn    = lane & 15;
  const int kend  = q0 + 16;            // causal bound (exclusive)

  for (int kp = 0; kp < kend; kp += 32) {
    // ---- scores S[16x32] = Q x K^T ----
    v8f s0 = splat8(0.0f), s1 = splat8(0.0f);
#pragma unroll
    for (int c = 0; c < 4; ++c) {
      s0 = wmma_bf16(qf[c], load_b_frag(Kbase + (size_t)(kp)      * HD + c * 32, HD, lane), s0);
      s1 = wmma_bf16(qf[c], load_b_frag(Kbase + (size_t)(kp + 16) * HD + c * 32, HD, lane), s1);
    }
    // ---- scale + causal mask ----
    const int col0 = kp + cn, col1 = col0 + 16;
#pragma unroll
    for (int r = 0; r < 8; ++r) {
      const int row = q0 + rbase + r;
      s0[r] = (col0 <= row) ? s0[r] * scale : -3.0e38f;
      s1[r] = (col1 <= row) ? s1[r] * scale : -3.0e38f;
    }
    // ---- online softmax ----
    v8f tm;
#pragma unroll
    for (int r = 0; r < 8; ++r) tm[r] = fmaxf(s0[r], s1[r]);
#pragma unroll
    for (int m = 1; m < 16; m <<= 1)
#pragma unroll
      for (int r = 0; r < 8; ++r) tm[r] = fmaxf(tm[r], __shfl_xor(tm[r], m, 32));

    v8f mnew, alpha;
#pragma unroll
    for (int r = 0; r < 8; ++r) {
      mnew[r]  = fmaxf(mi[r], tm[r]);
      alpha[r] = __expf(mi[r] - mnew[r]);
      s0[r]    = __expf(s0[r] - mnew[r]);
      s1[r]    = __expf(s1[r] - mnew[r]);
    }
    v8f rsum;
#pragma unroll
    for (int r = 0; r < 8; ++r) rsum[r] = s0[r] + s1[r];
#pragma unroll
    for (int m = 1; m < 16; m <<= 1)
#pragma unroll
      for (int r = 0; r < 8; ++r) rsum[r] += __shfl_xor(rsum[r], m, 32);
#pragma unroll
    for (int r = 0; r < 8; ++r) li[r] = li[r] * alpha[r] + rsum[r];
    mi = mnew;
#pragma unroll
    for (int t = 0; t < 8; ++t)
#pragma unroll
      for (int r = 0; r < 8; ++r) oacc[t][r] *= alpha[r];

    // ---- P: C layout -> A layout via LDS scratch ----
#pragma unroll
    for (int r = 0; r < 8; ++r) {
      scr[(rbase + r) * 32 + cn]      = (bf16)s0[r];
      scr[(rbase + r) * 32 + 16 + cn] = (bf16)s1[r];
    }
    __builtin_amdgcn_wave_barrier();   // LDS ops are in-order within a wave
    const v16bf pf = load_a_frag(scr, 32, lane);

    // ---- O += P x V (V pre-transposed: VT[d][s]) ----
#pragma unroll
    for (int t = 0; t < 8; ++t) {
      const v16bf vf = *(const v16bf*)(Vbase + (size_t)(t * 16 + cn) * SS + kp + ((lane >> 4) << 4));
      oacc[t] = wmma_bf16(pf, vf, oacc[t]);
    }
  }

  // ---- epilogue: O / l -> attnb[b][s][h*128 + d] (bf16) ----
  v8f linv;
#pragma unroll
  for (int r = 0; r < 8; ++r) linv[r] = 1.0f / li[r];
#pragma unroll
  for (int t = 0; t < 8; ++t)
#pragma unroll
    for (int r = 0; r < 8; ++r) {
      const int row = q0 + rbase + r;
      attnb[(size_t)(b * SS + row) * HH + h * HD + t * 16 + cn] =
          (bf16)(oacc[t][r] * linv[r]);
    }
}

// ---------------------------------------------------------------------------
// Host-side launch
// ---------------------------------------------------------------------------
extern "C" void kernel_launch(void* const* d_in, const int* in_sizes, int n_in,
                              void* d_out, int out_size, void* d_ws, size_t ws_size,
                              hipStream_t stream) {
  const float* hidden = (const float*)d_in[0];
  const int*   pos    = (const int*)  d_in[1];
  const float* wqkv   = (const float*)d_in[2];
  const float* qnw    = (const float*)d_in[3];
  const float* knw    = (const float*)d_in[4];
  const float* wo     = (const float*)d_in[5];
  float* out = (float*)d_out;

  // workspace carve-up (bytes)
  char* w = (char*)d_ws;
  bf16*  hb    = (bf16*)(w);                          // 4096*4096      bf16
  bf16*  wqkvb = (bf16*)(w + 33554432ull);            // 6144*4096      bf16
  bf16*  wob   = (bf16*)(w + 83886080ull);            // 4096*4096      bf16
  float* qkvf  = (float*)(w + 117440512ull);          // 4096*6144      f32
  bf16*  Qb    = (bf16*)(w + 218103808ull);           // 2*32*2048*128  bf16
  bf16*  Kbb   = (bf16*)(w + 251658240ull);           // 2*8*2048*128   bf16
  bf16*  VTb   = (bf16*)(w + 260046848ull);           // 2*8*128*2048   bf16
  bf16*  attnb = (bf16*)(w + 268435456ull);           // 4096*4096      bf16
  // total: 301989888 bytes

  // 1) converts
  {
    size_t n1 = (size_t)MTOT * HH;          // hidden
    size_t n2 = (size_t)QKVO * HH;          // wqkv
    size_t n3 = (size_t)HH * HH;            // wo
    f32_to_bf16_k<<<dim3((n1 + 255) / 256), dim3(256), 0, stream>>>(hidden, hb, n1);
    f32_to_bf16_k<<<dim3((n2 + 255) / 256), dim3(256), 0, stream>>>(wqkv, wqkvb, n2);
    f32_to_bf16_k<<<dim3((n3 + 255) / 256), dim3(256), 0, stream>>>(wo, wob, n3);
  }

  // 2) GEMM1: qkv = hidden x wqkv^T   [4096 x 6144]
  gemm_bf16_nt<<<dim3(QKVO / 256, MTOT / 128), dim3(256), 0, stream>>>(
      hb, wqkvb, qkvf, MTOT, QKVO, HH);

  // 3) RMSNorm + RoPE + split/transpose
  qkv_post_k<<<dim3(MTOT, NH + 2 * NKV), dim3(128), 0, stream>>>(
      qkvf, pos, qnw, knw, Qb, Kbb, VTb);

  // 4) flash attention
  flash_attn_k<<<dim3(SS / 64, BB * NH), dim3(128), 0, stream>>>(Qb, Kbb, VTb, attnb);

  // 5) GEMM2: out = attn x wo^T   [4096 x 4096], fp32 result
  gemm_bf16_nt<<<dim3(HH / 256, MTOT / 128), dim3(256), 0, stream>>>(
      attnb, wob, out, MTOT, HH, HH);
}
